// DeepSpeedSelfAttention_49143015801364
// MI455X (gfx1250) — matrix operand order repaired and checked by reading, not compiled, not run on hardware
//
#include <hip/hip_runtime.h>
#include <hip/hip_bf16.h>
#include <math.h>

// ---------------------------------------------------------------------------
// CDNA5 (gfx1250) fused DeepSpeed MLP block:
//   h  = input + residual + bias ; ln = LayerNorm(h)*nw + nb
//   inter = gelu_erf(ln @ inter_w + inter_b)        (bf16 WMMA GEMM)
//   out   = inter @ output_w + h + output_b         (bf16 WMMA GEMM)
// Matrix-core bound (~275 GFLOP vs ~0.8GB HBM traffic): everything routed
// through v_wmma_f32_16x16x32_bf16. 64x64 wave tiles (4x4 WMMA) give 4x
// fragment reuse. Tile staging uses gfx1250 async global->LDS copies
// (ASYNCcnt) when the toolchain exposes them; otherwise falls back to
// register-staged b128 loads + ds_store.
// ---------------------------------------------------------------------------

typedef __attribute__((ext_vector_type(16))) __bf16 v16bf;
typedef __attribute__((ext_vector_type(8)))  __bf16 v8bf;
typedef __attribute__((ext_vector_type(8)))  float  v8f;
typedef __attribute__((ext_vector_type(4)))  int    v4i;

#if __has_builtin(__builtin_amdgcn_global_load_async_to_lds_b128) && \
    __has_builtin(__builtin_amdgcn_s_wait_asynccnt)
#define USE_ASYNC_LDS 1
#else
#define USE_ASYNC_LDS 0
#endif

// ------------------------- fused residual + LayerNorm ----------------------
__global__ __launch_bounds__(256)
void fused_ln_kernel(const float* __restrict__ in, const float* __restrict__ res,
                     const float* __restrict__ bias, const float* __restrict__ nw,
                     const float* __restrict__ nb,
                     float* __restrict__ h, __bf16* __restrict__ ln, int H)
{
    const int row = blockIdx.x;
    const int tid = threadIdx.x;
    const size_t base = (size_t)row * H;

    __shared__ float s1[256];
    __shared__ float s2[256];

    float hv[4];
    float sum = 0.f, sq = 0.f;
#pragma unroll
    for (int i = 0; i < 4; ++i) {
        int idx = tid + i * 256;
        float x = in[base + idx] + res[base + idx] + bias[idx];
        hv[i] = x;
        sum += x;
        sq  += x * x;
    }
    s1[tid] = sum; s2[tid] = sq;
    __syncthreads();
#pragma unroll
    for (int off = 128; off > 0; off >>= 1) {
        if (tid < off) { s1[tid] += s1[tid + off]; s2[tid] += s2[tid + off]; }
        __syncthreads();
    }
    const float mu   = s1[0] / (float)H;
    const float var  = s2[0] / (float)H - mu * mu;
    const float rstd = rsqrtf(var + 1e-5f);

#pragma unroll
    for (int i = 0; i < 4; ++i) {
        int idx = tid + i * 256;
        float x = hv[i];
        h[base + idx] = x;
        float l = (x - mu) * rstd * nw[idx] + nb[idx];
        ln[base + idx] = (__bf16)l;
    }
}

// --------------------- f32 -> bf16 transpose (LDS tiled) -------------------
// W is R x C (row major, f32); Wt is C x R (row major, bf16): Wt[c][r]=W[r][c]
__global__ __launch_bounds__(256)
void transpose_f32_to_bf16(const float* __restrict__ W, __bf16* __restrict__ Wt,
                           int R, int C)
{
    __shared__ float t[32][33];
    const int c0 = blockIdx.x * 32;
    const int r0 = blockIdx.y * 32;
    const int tx = threadIdx.x, ty = threadIdx.y; // 32 x 8
#pragma unroll
    for (int i = 0; i < 4; ++i)
        t[ty + 8 * i][tx] = W[(size_t)(r0 + ty + 8 * i) * C + c0 + tx];
    __syncthreads();
#pragma unroll
    for (int i = 0; i < 4; ++i)
        Wt[(size_t)(c0 + ty + 8 * i) * R + r0 + tx] = (__bf16)t[tx][ty + 8 * i];
}

// ------------------------------ WMMA GEMM ----------------------------------
// C[M,N] = A[M,K] * Bt[N,K]^T   (A, Bt bf16 row-major)
// Block: 256 threads = 8 wave32s, 2(M) x 4(N); block tile 128 x 256.
// Wave tile: 64 x 64 = 4x4 WMMA 16x16 tiles -> 16 v_wmma per K-step,
// each A/B fragment reused 4x.
// EPI==0: out_bf[idx] = bf16(gelu_erf(acc + bias[col]))
// EPI==1: out_f [idx] = acc + hres[idx] + bias[col]
template <int EPI>
__global__ __launch_bounds__(256)
void gemm_wmma(const __bf16* __restrict__ A, const __bf16* __restrict__ Bt,
               int M, int N, int K,
               const float* __restrict__ bias, const float* __restrict__ hres,
               __bf16* __restrict__ out_bf, float* __restrict__ out_f)
{
    constexpr int BM = 128, BN = 256, TK = 32;
    constexpr int LDA = TK + 8;   // pad to 40 bf16 (80B) to spread LDS banks
    constexpr int LDB = TK + 8;

    __shared__ __bf16 sA[2][BM * LDA];   // 2 * 10240 B
    __shared__ __bf16 sB[2][BN * LDB];   // 2 * 20480 B  (total 60 KB)

    const int tid  = threadIdx.x;
    const int lane = tid & 31;
    const int wid  = tid >> 5;           // 8 wave32s
    const int wm   = wid & 1;            // 2 waves over M -> 64 rows each
    const int wn   = wid >> 1;           // 4 waves over N -> 64 cols each
    const int hl   = lane >> 4;          // half-wave selector
    const int l16  = lane & 15;

    const int m0 = blockIdx.x * BM;
    const int n0 = blockIdx.y * BN;

    auto load_tiles = [&](int buf, int k0) {
#if USE_ASYNC_LDS
        // gfx1250 async global->LDS: no VGPR staging, tracked by ASYNCcnt.
#pragma unroll
        for (int j = 0; j < 2; ++j) {
            int c  = tid + 256 * j;
            int r  = c >> 2;
            int cc = (c & 3) * 8;
            __builtin_amdgcn_global_load_async_to_lds_b128(
                (v4i*)&A[(size_t)(m0 + r) * K + k0 + cc],
                (v4i*)&sA[buf][r * LDA + cc],
                0, 0);
        }
#pragma unroll
        for (int j = 0; j < 4; ++j) {
            int c  = tid + 256 * j;
            int r  = c >> 2;
            int cc = (c & 3) * 8;
            __builtin_amdgcn_global_load_async_to_lds_b128(
                (v4i*)&Bt[(size_t)(n0 + r) * K + k0 + cc],
                (v4i*)&sB[buf][r * LDB + cc],
                0, 0);
        }
#else
        // A tile: 128 x 32 bf16 = 512 chunks of 8 bf16 (16B), 2 per thread
#pragma unroll
        for (int j = 0; j < 2; ++j) {
            int c  = tid + 256 * j;
            int r  = c >> 2;
            int cc = (c & 3) * 8;
            *(v8bf*)&sA[buf][r * LDA + cc] =
                *(const v8bf*)&A[(size_t)(m0 + r) * K + k0 + cc];
        }
        // B tile: 256 x 32 bf16 = 1024 chunks, 4 per thread
#pragma unroll
        for (int j = 0; j < 4; ++j) {
            int c  = tid + 256 * j;
            int r  = c >> 2;
            int cc = (c & 3) * 8;
            *(v8bf*)&sB[buf][r * LDB + cc] =
                *(const v8bf*)&Bt[(size_t)(n0 + r) * K + k0 + cc];
        }
#endif
    };

    auto async_fence = [&]() {
#if USE_ASYNC_LDS
        __builtin_amdgcn_s_wait_asynccnt(0);
#endif
    };

    v8f acc[4][4] = {};

    load_tiles(0, 0);
    async_fence();
    __syncthreads();

    const int nk = K / TK;
    for (int kt = 0; kt < nk; ++kt) {
        const int cur = kt & 1;
        if (kt + 1 < nk) load_tiles(cur ^ 1, (kt + 1) * TK);

        // A fragments: ISA layout — lanes 0-15: K 0-7 (v0-3) & 16-23 (v4-7),
        // lanes 16-31: K 8-15 & 24-31, row M = lane&15 (+tile offset)
        v16bf afr[4];
#pragma unroll
        for (int tm = 0; tm < 4; ++tm) {
            int row = wm * 64 + tm * 16 + l16;
            int kh  = hl * 8;
            union { v16bf v; v8bf h[2]; } u;
            u.h[0] = *(const v8bf*)&sA[cur][row * LDA + kh];
            u.h[1] = *(const v8bf*)&sA[cur][row * LDA + kh + 16];
            afr[tm] = u.v;
        }
        // B fragments: lanes 0-15 hold K 0-7 & 16-23 of column lane&15, lanes 16-31 K 8-15 & 24-31
        v16bf bfr[4];
#pragma unroll
        for (int tn = 0; tn < 4; ++tn) {
            int col = wn * 64 + tn * 16 + l16;
            int kb  = hl * 8;
            union { v16bf v; v8bf h[2]; } u;
            u.h[0] = *(const v8bf*)&sB[cur][col * LDB + kb];
            u.h[1] = *(const v8bf*)&sB[cur][col * LDB + kb + 16];
            bfr[tn] = u.v;
        }

#pragma unroll
        for (int tm = 0; tm < 4; ++tm)
#pragma unroll
            for (int tn = 0; tn < 4; ++tn)
                acc[tm][tn] = __builtin_amdgcn_wmma_f32_16x16x32_bf16(
                    false, afr[tm], false, bfr[tn],
                    (short)0, acc[tm][tn], false, false);

        async_fence();
        __syncthreads();
    }

    // Epilogue. C/D layout: N = lane&15, M = vgpr + 8*(lane>=16)
#pragma unroll
    for (int tm = 0; tm < 4; ++tm) {
        const int rowb = m0 + wm * 64 + tm * 16 + hl * 8;
#pragma unroll
        for (int tn = 0; tn < 4; ++tn) {
            const int col = n0 + wn * 64 + tn * 16 + l16;
            const float bcol = bias[col];
#pragma unroll
            for (int v = 0; v < 8; ++v) {
                const size_t idx = (size_t)(rowb + v) * N + col;
                float x = acc[tm][tn][v];
                if (EPI == 0) {
                    x += bcol;
                    float g = 0.5f * x * (1.0f + erff(x * 0.70710678118654752f));
                    out_bf[idx] = (__bf16)g;
                } else {
                    out_f[idx] = x + hres[idx] + bcol;
                }
            }
        }
    }
}

// ------------------------------- launcher ----------------------------------
extern "C" void kernel_launch(void* const* d_in, const int* in_sizes, int n_in,
                              void* d_out, int out_size, void* d_ws, size_t ws_size,
                              hipStream_t stream)
{
    const float* input    = (const float*)d_in[0];
    const float* residual = (const float*)d_in[1];
    const float* bias     = (const float*)d_in[2];
    const float* attn_nw  = (const float*)d_in[3];
    const float* attn_nb  = (const float*)d_in[4];
    const float* inter_w  = (const float*)d_in[5];
    const float* inter_b  = (const float*)d_in[6];
    const float* output_w = (const float*)d_in[7];
    const float* output_b = (const float*)d_in[8];
    float* out = (float*)d_out;

    const int H = in_sizes[2];           // 1024
    const int I = in_sizes[6];           // 4096
    const int M = in_sizes[0] / H;       // 16384 (= B*S)

    // Workspace layout (all 16B aligned):
    char* ws = (char*)d_ws;
    float*  h     = (float*)ws;                                   // M*H f32  (64 MB)
    __bf16* ln    = (__bf16*)(ws + (size_t)M * H * 4);            // M*H bf16 (32 MB)
    __bf16* wt1   = ln  + (size_t)M * H;                          // I*H bf16 ( 8 MB)
    __bf16* wt2   = wt1 + (size_t)I * H;                          // H*I bf16 ( 8 MB)
    __bf16* inter = wt2 + (size_t)H * I;                          // M*I bf16 (128 MB)

    // 1) fused residual+bias+LayerNorm -> h (f32), ln (bf16)
    fused_ln_kernel<<<M, 256, 0, stream>>>(input, residual, bias,
                                           attn_nw, attn_nb, h, ln, H);

    // 2) weight conversion + transpose: Wt[n][k] layouts for WMMA B fragments
    transpose_f32_to_bf16<<<dim3(I / 32, H / 32), dim3(32, 8), 0, stream>>>(
        inter_w, wt1, H, I);   // inter_w[H][I] -> wt1[I][H]
    transpose_f32_to_bf16<<<dim3(H / 32, I / 32), dim3(32, 8), 0, stream>>>(
        output_w, wt2, I, H);  // output_w[I][H] -> wt2[H][I]

    // 3) GEMM1: inter = gelu(ln @ inter_w + inter_b)   [M,I] bf16
    gemm_wmma<0><<<dim3(M / 128, I / 256), 256, 0, stream>>>(
        ln, wt1, M, I, H, inter_b, nullptr, inter, nullptr);

    // 4) GEMM2: out = inter @ output_w + h + output_b  [M,H] f32
    gemm_wmma<1><<<dim3(M / 128, H / 256), 256, 0, stream>>>(
        inter, wt2, M, H, I, output_b, h, nullptr, out);
}
